// Attention_730144440595
// MI455X (gfx1250) — matrix operand order repaired and checked by reading, not compiled, not run on hardware
//
#include <hip/hip_runtime.h>
#include <hip/hip_bf16.h>

// ---------------------------------------------------------------------------
// Types / helpers
// ---------------------------------------------------------------------------
typedef __attribute__((ext_vector_type(16))) __bf16 v16bf;
typedef __attribute__((ext_vector_type(8)))  float  v8f;

union BF16Frag { uint4 q[2]; v16bf v; };

__device__ __forceinline__ unsigned short f2bf(float f) {
    unsigned u = __float_as_uint(f);
    u += 0x7fffu + ((u >> 16) & 1u);         // round-to-nearest-even
    return (unsigned short)(u >> 16);
}
__device__ __forceinline__ float bf2f(unsigned short h) {
    return __uint_as_float(((unsigned)h) << 16);
}

// A-fragment (16x32 bf16): lane half h holds K = {8h..8h+7} U {16+8h..16+8h+7}
__device__ __forceinline__ v16bf load_a_frag(const unsigned short* row, int k0, int h) {
    BF16Frag f;
    f.q[0] = *(const uint4*)(row + k0 + h * 8);
    f.q[1] = *(const uint4*)(row + k0 + 16 + h * 8);
    return f.v;
}
// B-fragment (32x16 bf16, fed as N-major rows of B^T): lane half h holds K = {8h..8h+7} U {16+8h..16+8h+7}
__device__ __forceinline__ v16bf load_b_frag(const unsigned short* row, int k0, int h) {
    BF16Frag f;
    const uint4* p = (const uint4*)(row + k0 + h * 8);
    f.q[0] = p[0];
    f.q[1] = p[2];
    return f.v;
}

// CDNA5 async copy: 16 bytes per lane, global -> LDS, tracked by ASYNCcnt.
__device__ __forceinline__ void async_copy16(unsigned lds_off, const void* gptr) {
    asm volatile("global_load_async_to_lds_b128 %0, %1, off"
                 :: "v"(lds_off), "v"(gptr) : "memory");
}
__device__ __forceinline__ void wait_async0() {
    asm volatile("s_wait_asynccnt 0" ::: "memory");
}

#define WMMA_BF16(A, B, C) \
    __builtin_amdgcn_wmma_f32_16x16x32_bf16(false, (A), false, (B), (short)0, (C), false, false)

// Problem constants
#define BATCH 16
#define CH    256
#define NTOK  1024
#define HEADS 4
#define HDIM  64
#define QKVO  768
#define INNER 256

// padded LDS row strides (halfwords) to spread fragment loads across banks
#define BSH_LD 264   // 256 + 8   (528 B rows, 16B aligned)
#define KSH_LD 72    // 64 + 8    (144 B rows)
#define VSH_LD 40    // 32 + 8    (80 B rows)
#define PSH_LD 40    // 32 + 8

// ---------------------------------------------------------------------------
// Kernel 0: fp32 -> bf16 weight conversion
// ---------------------------------------------------------------------------
__global__ void cvt_kernel(const float* __restrict__ src, unsigned short* __restrict__ dst, int n) {
    int i = blockIdx.x * 256 + threadIdx.x;
    if (i < n) dst[i] = f2bf(src[i]);
}

// ---------------------------------------------------------------------------
// Kernel 1: channel LayerNorm -> token-major bf16 xnT[b, n, c]
// ---------------------------------------------------------------------------
__global__ void ln_kernel(const float* __restrict__ x, const float* __restrict__ g,
                          unsigned short* __restrict__ xnT) {
    __shared__ float stage[CH * 32];
    __shared__ float gsh[CH];
    __shared__ float ps[8][32], ps2[8][32];
    __shared__ float msh[32], rsh[32];

    const int tid = threadIdx.x;
    const int p0  = blockIdx.x * 32;
    const int b   = blockIdx.y;
    const float* xb = x + (size_t)b * CH * NTOK;

    for (int i = tid; i < CH * 32; i += 256) {
        int c = i >> 5, p = i & 31;
        stage[i] = xb[(size_t)c * NTOK + p0 + p];
    }
    gsh[tid] = g[tid];
    __syncthreads();

    const int pix = tid & 31, q = tid >> 5;
    float s = 0.f, s2 = 0.f;
    for (int c = q * 32; c < q * 32 + 32; ++c) {
        float v = stage[c * 32 + pix];
        s += v; s2 += v * v;
    }
    ps[q][pix] = s; ps2[q][pix] = s2;
    __syncthreads();

    if (tid < 32) {
        float ts = 0.f, t2 = 0.f;
        for (int qq = 0; qq < 8; ++qq) { ts += ps[qq][tid]; t2 += ps2[qq][tid]; }
        float mean = ts * (1.f / 256.f);
        float var  = t2 * (1.f / 256.f) - mean * mean;
        msh[tid] = mean;
        rsh[tid] = rsqrtf(var + 1e-5f);
    }
    __syncthreads();

    const float mean = msh[pix], rstd = rsh[pix];
    unsigned short* orow = xnT + ((size_t)b * NTOK + p0 + pix) * CH;
    for (int c = q * 32; c < q * 32 + 32; ++c)
        orow[c] = f2bf((stage[c * 32 + pix] - mean) * rstd * gsh[c]);
}

// ---------------------------------------------------------------------------
// Kernel 2: QKV GEMM  qkv[b, 768, n] = Wqkv[768,256] x xnT[b,n,256]^T
// grid (NTOK/16, 6, BATCH), block 256 = 8 waves, one 16x16 tile per wave.
// Shared B tile (16 tokens x 256 ch) async-staged to LDS once per block.
// ---------------------------------------------------------------------------
__global__ void gemm_qkv_kernel(const unsigned short* __restrict__ wq,
                                const unsigned short* __restrict__ xnT,
                                unsigned short* __restrict__ qkv) {
    __shared__ unsigned short bsh[16 * BSH_LD];

    const int tid = threadIdx.x;
    const int lane = tid & 31, wv = tid >> 5;
    const int h = lane >> 4, ln = lane & 15;
    const int ptile = blockIdx.x, b = blockIdx.z;
    const int otile = blockIdx.y * 8 + wv;

    // async stage: 16 rows x 256 halfwords = 512 chunks of 16 B
    {
        const unsigned short* bsrc = xnT + ((size_t)b * NTOK + ptile * 16) * CH;
        const unsigned base = (unsigned)(uintptr_t)&bsh[0];
        for (int e = tid; e < 512; e += 256) {
            int row = e >> 5, c16 = e & 31;
            async_copy16(base + (unsigned)(row * BSH_LD + c16 * 8) * 2,
                         bsrc + (size_t)row * CH + c16 * 8);
        }
    }
    wait_async0();
    __syncthreads();

    const unsigned short* arow = wq + (size_t)(otile * 16 + ln) * CH;
    const unsigned short* brow = &bsh[ln * BSH_LD];

    v8f acc = {};
    for (int k0 = 0; k0 < CH; k0 += 32) {
        __builtin_prefetch(arow + k0 + 64, 0, 0);
        v16bf a  = load_a_frag(arow, k0, h);
        v16bf bb = load_b_frag(brow, k0, h);
        acc = WMMA_BF16(a, bb, acc);
    }
    // D tile: lane = col N, rows M = h*8 + r
    unsigned short* out = qkv + (size_t)b * QKVO * NTOK
                        + (size_t)(otile * 16 + h * 8) * NTOK + ptile * 16 + ln;
    for (int r = 0; r < 8; ++r) out[(size_t)r * NTOK] = f2bf(acc[r]);
}

// ---------------------------------------------------------------------------
// Kernel 3: l2-normalize q,k per (head, token); transpose v -> vt[bh, d, n]
// ---------------------------------------------------------------------------
__global__ void normsplit_kernel(const unsigned short* __restrict__ qkv,
                                 unsigned short* __restrict__ qn,
                                 unsigned short* __restrict__ kn,
                                 unsigned short* __restrict__ vt) {
    const int p  = blockIdx.x * 256 + threadIdx.x;
    const int bh = blockIdx.y;
    const int b = bh >> 2, hh = bh & 3;
    const unsigned short* base = qkv + (size_t)b * QKVO * NTOK;

    {
        const unsigned short* src = base + (size_t)(hh * HDIM) * NTOK + p;
        float s = 0.f;
        for (int d = 0; d < HDIM; ++d) { float v = bf2f(src[(size_t)d * NTOK]); s += v * v; }
        float scale = 1.f / fmaxf(sqrtf(s), 1e-12f);
        unsigned short* dst = qn + ((size_t)bh * NTOK + p) * HDIM;
        for (int d = 0; d < HDIM; ++d) dst[d] = f2bf(bf2f(src[(size_t)d * NTOK]) * scale);
    }
    {
        const unsigned short* src = base + (size_t)(INNER + hh * HDIM) * NTOK + p;
        float s = 0.f;
        for (int d = 0; d < HDIM; ++d) { float v = bf2f(src[(size_t)d * NTOK]); s += v * v; }
        float scale = 1.f / fmaxf(sqrtf(s), 1e-12f);
        unsigned short* dst = kn + ((size_t)bh * NTOK + p) * HDIM;
        for (int d = 0; d < HDIM; ++d) dst[d] = f2bf(bf2f(src[(size_t)d * NTOK]) * scale);
    }
    {
        const unsigned short* src = base + (size_t)(2 * INNER + hh * HDIM) * NTOK + p;
        unsigned short* dst = vt + (size_t)bh * HDIM * NTOK + p;
        for (int d = 0; d < HDIM; ++d) dst[(size_t)d * NTOK] = src[(size_t)d * NTOK];
    }
}

// ---------------------------------------------------------------------------
// Kernel 4: flash attention per (b,h); block = 4 waves, each one 16-query tile.
// K/V chunks (32 keys) double-buffered in LDS via async DMA; the DMA of chunk
// i+1 overlaps the WMMA work of chunk i.
// grid (NTOK/64, BATCH*HEADS), block 128
// ---------------------------------------------------------------------------
__global__ void attn_kernel(const unsigned short* __restrict__ qn,
                            const unsigned short* __restrict__ kn,
                            const unsigned short* __restrict__ vt,
                            unsigned short* __restrict__ obn) {
    __shared__ unsigned short kbuf[2][32 * KSH_LD];   // 32 keys x 64 d (padded)
    __shared__ unsigned short vbuf[2][64 * VSH_LD];   // 64 d x 32 keys (padded)
    __shared__ unsigned short lds_p[4][16 * PSH_LD];  // per-wave P transpose

    const int tid = threadIdx.x;
    const int lane = tid & 31, wv = tid >> 5;
    const int h = lane >> 4, ln = lane & 15;
    const int bh = blockIdx.y;
    const int b = bh >> 2, hh = bh & 3;
    const int i0 = (blockIdx.x * 4 + wv) * 16;

    const unsigned short* qb = qn + (size_t)bh * NTOK * HDIM;
    const unsigned short* kb = kn + (size_t)bh * NTOK * HDIM;
    const unsigned short* vb = vt + (size_t)bh * HDIM * NTOK;

    const unsigned kbase[2] = { (unsigned)(uintptr_t)&kbuf[0][0],
                                (unsigned)(uintptr_t)&kbuf[1][0] };
    const unsigned vbase[2] = { (unsigned)(uintptr_t)&vbuf[0][0],
                                (unsigned)(uintptr_t)&vbuf[1][0] };

    auto issue_chunk = [&](int jt, int buf) {
        // K chunk: 32 rows x 64 hw -> 256 x 16B, 128 threads x 2
        for (int e = tid; e < 256; e += 128) {
            int row = e >> 3, c16 = e & 7;
            async_copy16(kbase[buf] + (unsigned)(row * KSH_LD + c16 * 8) * 2,
                         kb + (size_t)(jt + row) * HDIM + c16 * 8);
        }
        // V chunk: 64 rows x 32 hw -> 256 x 16B
        for (int e = tid; e < 256; e += 128) {
            int d = e >> 2, seg = e & 3;
            async_copy16(vbase[buf] + (unsigned)(d * VSH_LD + seg * 8) * 2,
                         vb + (size_t)d * NTOK + jt + seg * 8);
        }
    };

    const v16bf aq0 = load_a_frag(qb + (size_t)(i0 + ln) * HDIM, 0, h);
    const v16bf aq1 = load_a_frag(qb + (size_t)(i0 + ln) * HDIM, 32, h);

    v8f o0 = {}, o1 = {}, o2 = {}, o3 = {};
    float m[8], l[8], alpha[8];
    for (int r = 0; r < 8; ++r) { m[r] = -1e30f; l[r] = 0.f; }

    unsigned short* pl = &lds_p[wv][0];

    issue_chunk(0, 0);
    for (int c = 0; c < NTOK / 32; ++c) {
        const int buf = c & 1;
        wait_async0();          // my share of chunk c has landed
        __syncthreads();        // everyone's share landed; prev compute done
        if (c + 1 < NTOK / 32) issue_chunk((c + 1) * 32, buf ^ 1);

        const unsigned short* kc = &kbuf[buf][0];
        const unsigned short* vc = &vbuf[buf][0];

        // --- scores S = (Q K^T) * 8 for 16 queries x 32 keys (K from LDS) ---
        v8f s0 = {}, s1 = {};
        {
            v16bf bk;
            bk = load_b_frag(kc + ln * KSH_LD, 0, h);         s0 = WMMA_BF16(aq0, bk, s0);
            bk = load_b_frag(kc + ln * KSH_LD, 32, h);        s0 = WMMA_BF16(aq1, bk, s0);
            bk = load_b_frag(kc + (16 + ln) * KSH_LD, 0, h);  s1 = WMMA_BF16(aq0, bk, s1);
            bk = load_b_frag(kc + (16 + ln) * KSH_LD, 32, h); s1 = WMMA_BF16(aq1, bk, s1);
        }
        // --- online softmax (rows live across 16-lane halves) ---
        for (int r = 0; r < 8; ++r) {
            float a0 = s0[r] * 8.0f, a1 = s1[r] * 8.0f;
            float rm = fmaxf(a0, a1);
            for (int off = 1; off < 16; off <<= 1) rm = fmaxf(rm, __shfl_xor(rm, off, 32));
            float mn = fmaxf(m[r], rm);
            float al = __expf(m[r] - mn);
            float p0 = __expf(a0 - mn), p1 = __expf(a1 - mn);
            float rs = p0 + p1;
            for (int off = 1; off < 16; off <<= 1) rs += __shfl_xor(rs, off, 32);
            l[r] = l[r] * al + rs;
            m[r] = mn;
            alpha[r] = al;
            int row = h * 8 + r;
            pl[row * PSH_LD + ln]      = f2bf(p0);
            pl[row * PSH_LD + 16 + ln] = f2bf(p1);
        }
        asm volatile("s_wait_dscnt 0" ::: "memory");  // wave-local LDS store->load
        for (int r = 0; r < 8; ++r) {
            o0[r] *= alpha[r]; o1[r] *= alpha[r]; o2[r] *= alpha[r]; o3[r] *= alpha[r];
        }
        // --- O += P V (P from LDS A-layout; V^T rows from LDS) ---
        v16bf pa = load_a_frag(pl + ln * PSH_LD, 0, h);
        v16bf bv;
        bv = load_b_frag(vc + (0 * 16 + ln) * VSH_LD, 0, h); o0 = WMMA_BF16(pa, bv, o0);
        bv = load_b_frag(vc + (1 * 16 + ln) * VSH_LD, 0, h); o1 = WMMA_BF16(pa, bv, o1);
        bv = load_b_frag(vc + (2 * 16 + ln) * VSH_LD, 0, h); o2 = WMMA_BF16(pa, bv, o2);
        bv = load_b_frag(vc + (3 * 16 + ln) * VSH_LD, 0, h); o3 = WMMA_BF16(pa, bv, o3);
    }

    // --- finalize: divide by row-sum, store token-major obn[b, n, inner] ---
    unsigned short* ob = obn + (size_t)b * NTOK * INNER;
    for (int r = 0; r < 8; ++r) {
        float inv = 1.f / l[r];
        int tok = i0 + h * 8 + r;
        unsigned short* orow = ob + (size_t)tok * INNER + hh * HDIM;
        orow[0 * 16 + ln] = f2bf(o0[r] * inv);
        orow[1 * 16 + ln] = f2bf(o1[r] * inv);
        orow[2 * 16 + ln] = f2bf(o2[r] * inv);
        orow[3 * 16 + ln] = f2bf(o3[r] * inv);
    }
}

// ---------------------------------------------------------------------------
// Kernel 5: output projection + residual: y[b,o,p] = Wout x O^T + x
// grid (NTOK/16, 2, BATCH), block 256 = 8 waves; shared B tile via async LDS
// ---------------------------------------------------------------------------
__global__ void proj_kernel(const unsigned short* __restrict__ wo,
                            const unsigned short* __restrict__ obn,
                            const float* __restrict__ x,
                            float* __restrict__ y) {
    __shared__ unsigned short bsh[16 * BSH_LD];

    const int tid = threadIdx.x;
    const int lane = tid & 31, wv = tid >> 5;
    const int h = lane >> 4, ln = lane & 15;
    const int ptile = blockIdx.x, b = blockIdx.z;
    const int otile = blockIdx.y * 8 + wv;

    {
        const unsigned short* bsrc = obn + ((size_t)b * NTOK + ptile * 16) * INNER;
        const unsigned base = (unsigned)(uintptr_t)&bsh[0];
        for (int e = tid; e < 512; e += 256) {
            int row = e >> 5, c16 = e & 31;
            async_copy16(base + (unsigned)(row * BSH_LD + c16 * 8) * 2,
                         bsrc + (size_t)row * INNER + c16 * 8);
        }
    }
    wait_async0();
    __syncthreads();

    const unsigned short* arow = wo + (size_t)(otile * 16 + ln) * INNER;
    const unsigned short* brow = &bsh[ln * BSH_LD];

    v8f acc = {};
    for (int k0 = 0; k0 < INNER; k0 += 32) {
        __builtin_prefetch(arow + k0 + 64, 0, 0);
        v16bf a  = load_a_frag(arow, k0, h);
        v16bf bb = load_b_frag(brow, k0, h);
        acc = WMMA_BF16(a, bb, acc);
    }
    const size_t base = (size_t)b * CH * NTOK + (size_t)(otile * 16 + h * 8) * NTOK
                      + ptile * 16 + ln;
    for (int r = 0; r < 8; ++r)
        y[base + (size_t)r * NTOK] = acc[r] + x[base + (size_t)r * NTOK];
}

// ---------------------------------------------------------------------------
// Launch
// ---------------------------------------------------------------------------
extern "C" void kernel_launch(void* const* d_in, const int* in_sizes, int n_in,
                              void* d_out, int out_size, void* d_ws, size_t ws_size,
                              hipStream_t stream) {
    const float* x    = (const float*)d_in[0];
    const float* g    = (const float*)d_in[1];
    const float* wqkv = (const float*)d_in[2];
    const float* wout = (const float*)d_in[3];
    float* y = (float*)d_out;

    auto align256 = [](size_t v) { return (v + 255) & ~(size_t)255; };
    char* w = (char*)d_ws;
    unsigned short* wq_bf = (unsigned short*)w; w += align256((size_t)QKVO * CH * 2);
    unsigned short* wo_bf = (unsigned short*)w; w += align256((size_t)CH * INNER * 2);
    unsigned short* xnT   = (unsigned short*)w; w += align256((size_t)BATCH * NTOK * CH * 2);
    unsigned short* qkv   = (unsigned short*)w; w += align256((size_t)BATCH * QKVO * NTOK * 2);
    unsigned short* qn    = (unsigned short*)w; w += align256((size_t)BATCH * HEADS * NTOK * HDIM * 2);
    unsigned short* kn    = (unsigned short*)w; w += align256((size_t)BATCH * HEADS * NTOK * HDIM * 2);
    unsigned short* vt    = (unsigned short*)w; w += align256((size_t)BATCH * HEADS * HDIM * NTOK * 2);
    unsigned short* obn   = (unsigned short*)w; w += align256((size_t)BATCH * NTOK * INNER * 2);

    cvt_kernel<<<(QKVO * CH + 255) / 256, 256, 0, stream>>>(wqkv, wq_bf, QKVO * CH);
    cvt_kernel<<<(CH * INNER + 255) / 256, 256, 0, stream>>>(wout, wo_bf, CH * INNER);
    ln_kernel<<<dim3(NTOK / 32, BATCH), 256, 0, stream>>>(x, g, xnT);
    gemm_qkv_kernel<<<dim3(NTOK / 16, QKVO / 128, BATCH), 256, 0, stream>>>(wq_bf, xnT, qkv);
    normsplit_kernel<<<dim3(NTOK / 256, BATCH * HEADS), 256, 0, stream>>>(qkv, qn, kn, vt);
    attn_kernel<<<dim3(NTOK / 64, BATCH * HEADS), 128, 0, stream>>>(qn, kn, vt, obn);
    proj_kernel<<<dim3(NTOK / 16, CH / 128, BATCH), 256, 0, stream>>>(wo_bf, obn, x, y);
}
